// Transformer_55679956025707
// MI455X (gfx1250) — matrix operand order repaired and checked by reading, not compiled, not run on hardware
//
#include <hip/hip_runtime.h>
#include <math.h>

#define Bq 8
#define Tt 1024
#define Cc 1024
#define NHh 16
#define HDd 64
#define NLl 8
#define Mrows (Bq * Tt)   // 8192

typedef __attribute__((ext_vector_type(16))) __bf16        v16bf;
typedef __attribute__((ext_vector_type(8)))  float         v8f;
typedef __attribute__((ext_vector_type(4)))  unsigned int  u32x4;

union Frag16 {
    v16bf v;
    u32x4 q[2];
    unsigned short s[16];
};

// fp32 -> bf16 via the native conversion path (RNE in hardware)
__device__ __forceinline__ unsigned short f2bf(float f) {
    union { __bf16 b; unsigned short u; } cv;
    cv.b = (__bf16)f;
    return cv.u;
}
// packed pair: lo in [15:0], hi in [31:16] (targets v_cvt_pk_bf16_f32)
__device__ __forceinline__ unsigned int f2bf2(float lo, float hi) {
    union { __bf16 b[2]; unsigned int u; } cv;
    cv.b[0] = (__bf16)lo;
    cv.b[1] = (__bf16)hi;
    return cv.u;
}

// ---------------------------------------------------------------------------
// residual-stream init
// ---------------------------------------------------------------------------
__global__ void copy_kernel(const float* __restrict__ a, float* __restrict__ b, size_t n) {
    for (size_t i = (size_t)blockIdx.x * blockDim.x + threadIdx.x; i < n;
         i += (size_t)gridDim.x * blockDim.x)
        b[i] = a[i];
}

// ---------------------------------------------------------------------------
// LayerNorm: fp32 row -> bf16 (obf) or fp32 (of32). One block per row, C=1024.
// ---------------------------------------------------------------------------
__global__ __launch_bounds__(256) void ln_kernel(
    const float* __restrict__ x, const float* __restrict__ w, const float* __restrict__ b,
    unsigned short* __restrict__ obf, float* __restrict__ of32)
{
    const int row = blockIdx.x;
    const int t   = threadIdx.x;
    __shared__ float red[256];

    const float4 xv = *(const float4*)(x + (size_t)row * Cc + t * 4);
    red[t] = xv.x + xv.y + xv.z + xv.w;
    __syncthreads();
    for (int o = 128; o > 0; o >>= 1) { if (t < o) red[t] += red[t + o]; __syncthreads(); }
    const float mean = red[0] * (1.0f / Cc);
    __syncthreads();

    const float dx = xv.x - mean, dy = xv.y - mean, dz = xv.z - mean, dw = xv.w - mean;
    red[t] = dx * dx + dy * dy + dz * dz + dw * dw;
    __syncthreads();
    for (int o = 128; o > 0; o >>= 1) { if (t < o) red[t] += red[t + o]; __syncthreads(); }
    const float rstd = rsqrtf(red[0] * (1.0f / Cc) + 1e-5f);

    const float4 wv = *(const float4*)(w + t * 4);
    const float4 bv = *(const float4*)(b + t * 4);
    const float o0 = dx * rstd * wv.x + bv.x;
    const float o1 = dy * rstd * wv.y + bv.y;
    const float o2 = dz * rstd * wv.z + bv.z;
    const float o3 = dw * rstd * wv.w + bv.w;

    const size_t base = (size_t)row * Cc + t * 4;
    if (obf) {
        unsigned int* o32 = (unsigned int*)(obf + base);
        o32[0] = f2bf2(o0, o1);
        o32[1] = f2bf2(o2, o3);
    } else {
        of32[base + 0] = o0; of32[base + 1] = o1;
        of32[base + 2] = o2; of32[base + 3] = o3;
    }
}

// ---------------------------------------------------------------------------
// GEMM: Y[M,N] = X[M,K](bf16) * W[N,K]^T(f32->bf16 on the fly) + bias
//   mode 0: out bf16    mode 1: resid[idx] += val (fp32)    mode 2: GELU -> bf16
// Block tile 128x256, 8 waves each 64x64 (16 accumulators), BK=32,
// double-buffered LDS + register-staged global loads (1 barrier per K-step).
// ---------------------------------------------------------------------------
#define BM 128
#define BN 256
#define BK 32

__global__ __launch_bounds__(256) void gemm_bf16(
    const unsigned short* __restrict__ A,   // [M,K] bf16
    const float* __restrict__ W,            // [N,K] f32 (torch [out,in])
    const float* __restrict__ bias,         // [N]
    unsigned short* __restrict__ outbf,     // modes 0/2
    float* __restrict__ resid,              // mode 1 (in/out)
    int Ndim, int K, int mode)
{
    __shared__ unsigned short sA[2][BM][BK];   // 16 KB
    __shared__ unsigned short sW[2][BN][BK];   // 32 KB

    const int tid  = threadIdx.x;
    const int lane = tid & 31;
    const int wave = tid >> 5;
    const int lm   = lane & 15;
    const int lh   = lane >> 4;            // lane half
    const int wm   = (wave & 1) * 64;      // wave row offset in block tile
    const int wn   = (wave >> 1) * 64;     // wave col offset in block tile
    const int m0   = blockIdx.y * BM;
    const int n0   = blockIdx.x * BN;
    const int arow = tid >> 1, ach = tid & 1;

    v8f acc[4][4];
    #pragma unroll
    for (int mt = 0; mt < 4; ++mt)
        #pragma unroll
        for (int nt = 0; nt < 4; ++nt)
            acc[mt][nt] = (v8f){0.f, 0.f, 0.f, 0.f, 0.f, 0.f, 0.f, 0.f};

    {   // prologue: stage K-step 0 into buffer 0
        const u32x4* asrc = (const u32x4*)(A + (size_t)(m0 + arow) * K + ach * 16);
        u32x4* adst = (u32x4*)&sA[0][arow][ach * 16];
        adst[0] = asrc[0]; adst[1] = asrc[1];

        const float4* wsrc = (const float4*)(W + (size_t)(n0 + tid) * K);
        union { unsigned int u[8]; u32x4 q[2]; } wr;
        #pragma unroll
        for (int j = 0; j < 4; ++j) {
            const float4 f = wsrc[j];
            wr.u[j * 2 + 0] = f2bf2(f.x, f.y);
            wr.u[j * 2 + 1] = f2bf2(f.z, f.w);
        }
        u32x4* wdst = (u32x4*)&sW[0][tid][0];
        wdst[0] = wr.q[0]; wdst[1] = wr.q[1];
        #pragma unroll
        for (int j = 0; j < 4; ++j) {
            const float4 f = wsrc[4 + j];
            wr.u[j * 2 + 0] = f2bf2(f.x, f.y);
            wr.u[j * 2 + 1] = f2bf2(f.z, f.w);
        }
        wdst[2] = wr.q[0]; wdst[3] = wr.q[1];
    }
    __syncthreads();

    const int nk = K / BK;
    for (int kk = 0; kk < nk; ++kk) {
        const int cur = kk & 1;
        const bool more = (kk + 1 < nk);

        // 1) issue next tile's global loads into registers (overlaps WMMA chain)
        u32x4 an0, an1;
        float4 wf[8];
        if (more) {
            const int k0n = (kk + 1) * BK;
            const u32x4* asrc = (const u32x4*)(A + (size_t)(m0 + arow) * K + k0n + ach * 16);
            an0 = asrc[0]; an1 = asrc[1];
            const float4* wsrc = (const float4*)(W + (size_t)(n0 + tid) * K + k0n);
            #pragma unroll
            for (int j = 0; j < 8; ++j) wf[j] = wsrc[j];
        }

        // 2) compute on current buffer: 16 WMMAs per wave
        Frag16 af[4];
        #pragma unroll
        for (int mt = 0; mt < 4; ++mt) {
            // 16-bit A 16x32 layout: half0 lanes hold K {0..7,16..23}, half1 {8..15,24..31}
            af[mt].q[0] = *(const u32x4*)&sA[cur][wm + mt * 16 + lm][lh * 8];
            af[mt].q[1] = *(const u32x4*)&sA[cur][wm + mt * 16 + lm][16 + lh * 8];
        }
        #pragma unroll
        for (int nt = 0; nt < 4; ++nt) {
            Frag16 bf;   // B 32x16 layout: half0 lanes hold K {0..7,16..23}, half1 {8..15,24..31}
            bf.q[0] = *(const u32x4*)&sW[cur][wn + nt * 16 + lm][lh * 8];
            bf.q[1] = *(const u32x4*)&sW[cur][wn + nt * 16 + lm][16 + lh * 8];
            #pragma unroll
            for (int mt = 0; mt < 4; ++mt)
                acc[mt][nt] = __builtin_amdgcn_wmma_f32_16x16x32_bf16(
                    false, af[mt].v, false, bf.v, (short)0, acc[mt][nt], false, false);
        }

        // 3) convert + store staged tile into the other buffer
        if (more) {
            const int nb = cur ^ 1;
            u32x4* adst = (u32x4*)&sA[nb][arow][ach * 16];
            adst[0] = an0; adst[1] = an1;
            union { unsigned int u[16]; u32x4 q[4]; } wr;
            #pragma unroll
            for (int j = 0; j < 8; ++j) {
                wr.u[j * 2 + 0] = f2bf2(wf[j].x, wf[j].y);
                wr.u[j * 2 + 1] = f2bf2(wf[j].z, wf[j].w);
            }
            u32x4* wdst = (u32x4*)&sW[nb][tid][0];
            wdst[0] = wr.q[0]; wdst[1] = wr.q[1]; wdst[2] = wr.q[2]; wdst[3] = wr.q[3];
        }
        __syncthreads();
    }

    // epilogue — C/D layout: VGPR r -> row r (+8 for upper lane half), col = lane&15
    #pragma unroll
    for (int mt = 0; mt < 4; ++mt) {
        #pragma unroll
        for (int nt = 0; nt < 4; ++nt) {
            const int col = n0 + wn + nt * 16 + lm;
            const float bval = bias[col];
            #pragma unroll
            for (int r = 0; r < 8; ++r) {
                const int row = m0 + wm + mt * 16 + r + lh * 8;
                const size_t idx = (size_t)row * Ndim + col;
                const float val = acc[mt][nt][r] + bval;
                if (mode == 0) {
                    outbf[idx] = f2bf(val);
                } else if (mode == 1) {
                    resid[idx] += val;
                } else {
                    const float g = 0.5f * val * (1.0f + erff(val * 0.70710678f));
                    outbf[idx] = f2bf(g);
                }
            }
        }
    }
}

// ---------------------------------------------------------------------------
// Flash attention, block-causal. mask == (j < (i/64+1)*64) so 64-aligned tiles
// need no masking. One block per (batch, head, 64-query tile); 8 waves.
// ---------------------------------------------------------------------------
__global__ __launch_bounds__(256) void attn_kernel(
    const unsigned short* __restrict__ Q,
    const unsigned short* __restrict__ K,
    const unsigned short* __restrict__ V,
    unsigned short* __restrict__ Y)
{
    const int qb = blockIdx.x & 15;
    const int hh = (blockIdx.x >> 4) & (NHh - 1);
    const int bb = blockIdx.x >> 8;
    const int t    = threadIdx.x;
    const int lane = t & 31;
    const int wave = t >> 5;
    const int lm   = lane & 15;
    const int lh   = lane >> 4;
    const int mt    = wave & 3;            // 16-row strip of the 64x64 tile
    const int ncol0 = (wave >> 2) * 32;    // 32-col strip

    __shared__ unsigned short sQ[64][64];
    __shared__ unsigned short sK[64][64];
    __shared__ unsigned short sV[64][64];
    __shared__ float sS[64 * 64];          // f32 scores; reused (overlaid) as bf16 P
    __shared__ float sO[64][64];
    __shared__ float sm[64], sl[64];
    unsigned short* sP = (unsigned short*)sS;

    {   // load Q tile (64x64 bf16)
        const int r = t >> 2, ch = t & 3;
        const u32x4* src = (const u32x4*)(Q + (size_t)(bb * Tt + qb * 64 + r) * Cc + hh * HDd + ch * 16);
        u32x4* dst = (u32x4*)&sQ[r][ch * 16];
        dst[0] = src[0]; dst[1] = src[1];
    }
    if (t < 64) { sm[t] = -1e30f; sl[t] = 0.f; }
    {
        float* of = &sO[0][0];
        #pragma unroll
        for (int i = 0; i < 16; ++i) of[t * 16 + i] = 0.f;
    }

    float p[64];   // per-thread softmax row (threads 0..63), survives the barrier

    for (int kt = 0; kt <= qb; ++kt) {
        __syncthreads();   // prev iteration's PV reads of sV done
        {   // load K,V tiles
            const int r = t >> 2, ch = t & 3;
            const size_t rowb = (size_t)(bb * Tt + kt * 64 + r) * Cc + hh * HDd + ch * 16;
            const u32x4* ks = (const u32x4*)(K + rowb);
            const u32x4* vs = (const u32x4*)(V + rowb);
            u32x4* kd = (u32x4*)&sK[r][ch * 16];
            u32x4* vd = (u32x4*)&sV[r][ch * 16];
            kd[0] = ks[0]; kd[1] = ks[1];
            vd[0] = vs[0]; vd[1] = vs[1];
        }
        __syncthreads();

        {   // S = Q K^T  (K-dim = HD = 64 -> two WMMAs per 16x16 tile)
            Frag16 a0, a1;
            a0.q[0] = *(const u32x4*)&sQ[mt * 16 + lm][lh * 8];
            a0.q[1] = *(const u32x4*)&sQ[mt * 16 + lm][16 + lh * 8];
            a1.q[0] = *(const u32x4*)&sQ[mt * 16 + lm][32 + lh * 8];
            a1.q[1] = *(const u32x4*)&sQ[mt * 16 + lm][48 + lh * 8];
            #pragma unroll
            for (int nt = 0; nt < 2; ++nt) {
                const int n = ncol0 + nt * 16 + lm;      // key row
                Frag16 b0, b1;
                b0.q[0] = *(const u32x4*)&sK[n][lh * 8];
                b0.q[1] = *(const u32x4*)&sK[n][16 + lh * 8];
                b1.q[0] = *(const u32x4*)&sK[n][32 + lh * 8];
                b1.q[1] = *(const u32x4*)&sK[n][48 + lh * 8];
                v8f sc = (v8f){0.f, 0.f, 0.f, 0.f, 0.f, 0.f, 0.f, 0.f};
                sc = __builtin_amdgcn_wmma_f32_16x16x32_bf16(false, a0.v, false, b0.v, (short)0, sc, false, false);
                sc = __builtin_amdgcn_wmma_f32_16x16x32_bf16(false, a1.v, false, b1.v, (short)0, sc, false, false);
                #pragma unroll
                for (int r = 0; r < 8; ++r)
                    sS[(mt * 16 + r + lh * 8) * 64 + ncol0 + nt * 16 + lm] = sc[r];
            }
        }
        __syncthreads();

        if (t < 64) {   // online softmax, one row per thread
            const float mprev = sm[t];
            float mx = mprev;
            #pragma unroll 8
            for (int c = 0; c < 64; ++c) { p[c] = sS[t * 64 + c] * 0.125f; mx = fmaxf(mx, p[c]); }
            const float f = __expf(mprev - mx);
            float sum = 0.f;
            #pragma unroll 8
            for (int c = 0; c < 64; ++c) { p[c] = __expf(p[c] - mx); sum += p[c]; }
            sl[t] = sl[t] * f + sum;
            sm[t] = mx;
            #pragma unroll 8
            for (int c = 0; c < 64; ++c) sO[t][c] *= f;
        }
        __syncthreads();   // all S reads done before P overlays it
        if (t < 64) {
            unsigned int* p32 = (unsigned int*)(sP + t * 64);
            #pragma unroll 8
            for (int c = 0; c < 32; ++c) p32[c] = f2bf2(p[2 * c], p[2 * c + 1]);
        }
        __syncthreads();

        {   // O += P V   (K-dim = 64 keys -> two WMMAs per tile)
            Frag16 a0, a1;
            a0.q[0] = *(const u32x4*)&sP[(mt * 16 + lm) * 64 + lh * 8];
            a0.q[1] = *(const u32x4*)&sP[(mt * 16 + lm) * 64 + 16 + lh * 8];
            a1.q[0] = *(const u32x4*)&sP[(mt * 16 + lm) * 64 + 32 + lh * 8];
            a1.q[1] = *(const u32x4*)&sP[(mt * 16 + lm) * 64 + 48 + lh * 8];
            #pragma unroll
            for (int nt = 0; nt < 2; ++nt) {
                const int d = ncol0 + nt * 16 + lm;
                Frag16 b0, b1;
                #pragma unroll
                for (int e = 0; e < 16; ++e) {        // B[k][n] = V[key][d], column gather
                    b0.s[e] = sV[lh * 8 + (e & 7) + (e >> 3) * 16][d];
                    b1.s[e] = sV[32 + lh * 8 + (e & 7) + (e >> 3) * 16][d];
                }
                v8f c;
                #pragma unroll
                for (int r = 0; r < 8; ++r) c[r] = sO[mt * 16 + r + lh * 8][d];
                c = __builtin_amdgcn_wmma_f32_16x16x32_bf16(false, a0.v, false, b0.v, (short)0, c, false, false);
                c = __builtin_amdgcn_wmma_f32_16x16x32_bf16(false, a1.v, false, b1.v, (short)0, c, false, false);
                #pragma unroll
                for (int r = 0; r < 8; ++r) sO[mt * 16 + r + lh * 8][d] = c[r];
            }
        }
    }
    __syncthreads();

    #pragma unroll
    for (int i = 0; i < 16; ++i) {     // normalize + store bf16
        const int idx = t * 16 + i;
        const int r = idx >> 6, d = idx & 63;
        Y[(size_t)(bb * Tt + qb * 64 + r) * Cc + hh * HDd + d] = f2bf(sO[r][d] / sl[r]);
    }
}

// ---------------------------------------------------------------------------
extern "C" void kernel_launch(void* const* d_in, const int* in_sizes, int n_in,
                              void* d_out, int out_size, void* d_ws, size_t ws_size,
                              hipStream_t stream)
{
    (void)in_sizes; (void)n_in; (void)out_size; (void)ws_size;

    const float* seq  = (const float*)d_in[0];
    // d_in[1] = mask: unused — block-causal mask reduces to j < (i/64+1)*64
    const float* Wq   = (const float*)d_in[2];
    const float* bq   = (const float*)d_in[3];
    const float* Wk   = (const float*)d_in[4];
    const float* bk   = (const float*)d_in[5];
    const float* Wv   = (const float*)d_in[6];
    const float* bv   = (const float*)d_in[7];
    const float* Wo   = (const float*)d_in[8];
    const float* bo   = (const float*)d_in[9];
    const float* ln1w = (const float*)d_in[10];
    const float* ln1b = (const float*)d_in[11];
    const float* ln2w = (const float*)d_in[12];
    const float* ln2b = (const float*)d_in[13];
    const float* W1   = (const float*)d_in[14];
    const float* b1   = (const float*)d_in[15];
    const float* W2   = (const float*)d_in[16];
    const float* b2   = (const float*)d_in[17];
    const float* lnfw = (const float*)d_in[18];
    const float* lnfb = (const float*)d_in[19];

    char* ws = (char*)d_ws;
    float*          x    = (float*)ws;                                      // M*C f32
    unsigned short* h    = (unsigned short*)(ws + (size_t)Mrows * Cc * 4);  // M*C bf16
    unsigned short* qbuf = h + (size_t)Mrows * Cc;
    unsigned short* kbuf = qbuf + (size_t)Mrows * Cc;
    unsigned short* vbuf = kbuf + (size_t)Mrows * Cc;
    unsigned short* ybuf = vbuf + (size_t)Mrows * Cc;
    unsigned short* hid  = qbuf;   // overlays q/k/v/y: M * 4C bf16 (same footprint)

    copy_kernel<<<1024, 256, 0, stream>>>(seq, x, (size_t)Mrows * Cc);

    const dim3 gC (Cc / BN,     Mrows / BM);    // N=1024 GEMMs -> (4, 64)
    const dim3 g4C(4 * Cc / BN, Mrows / BM);    // N=4096 GEMM  -> (16, 64)

    for (int i = 0; i < NLl; ++i) {
        const size_t wOff = (size_t)i * Cc * Cc;
        ln_kernel<<<Mrows, 256, 0, stream>>>(x, ln1w + i * Cc, ln1b + i * Cc, h, nullptr);
        gemm_bf16<<<gC, 256, 0, stream>>>(h, Wq + wOff, bq + i * Cc, qbuf, nullptr, Cc, Cc, 0);
        gemm_bf16<<<gC, 256, 0, stream>>>(h, Wk + wOff, bk + i * Cc, kbuf, nullptr, Cc, Cc, 0);
        gemm_bf16<<<gC, 256, 0, stream>>>(h, Wv + wOff, bv + i * Cc, vbuf, nullptr, Cc, Cc, 0);
        attn_kernel<<<Bq * NHh * (Tt / 64), 256, 0, stream>>>(qbuf, kbuf, vbuf, ybuf);
        gemm_bf16<<<gC, 256, 0, stream>>>(ybuf, Wo + wOff, bo + i * Cc, nullptr, x, Cc, Cc, 1);
        ln_kernel<<<Mrows, 256, 0, stream>>>(x, ln2w + i * Cc, ln2b + i * Cc, h, nullptr);
        gemm_bf16<<<g4C, 256, 0, stream>>>(h,  W1 + (size_t)i * 4 * Cc * Cc, b1 + (size_t)i * 4 * Cc,
                                           hid, nullptr, 4 * Cc, Cc, 2);
        gemm_bf16<<<gC, 256, 0, stream>>>(hid, W2 + (size_t)i * Cc * 4 * Cc, b2 + i * Cc,
                                          nullptr, x, Cc, 4 * Cc, 1);
    }
    ln_kernel<<<Mrows, 256, 0, stream>>>(x, lnfw, lnfb, nullptr, (float*)d_out);
}
